// QuasiLSTMlayer_6193342841506
// MI455X (gfx1250) — matrix operand order repaired and checked by reading, not compiled, not run on hardware
//
#include <hip/hip_runtime.h>
#include <hip/hip_bf16.h>

// ---------------------------------------------------------------------------
// QuasiLSTM layer for MI455X (gfx1250, wave32, WMMA).
//   x:      [1024, 32, 1024] f32   (slen, bsz, in_dim) -> flat M = 32768 rows
//   wm_z/f: [1024, 1024] f32       (hidden, in) -> used as B matrix (K-major)
//   out_z/f = x @ wm^T  (bf16 WMMA, f32 accumulate)
//   scan:   diagonal recurrence per (b,h) channel over 1024 steps
// ---------------------------------------------------------------------------

#define SLEN    1024
#define BSZ     32
#define IN_DIM  1024
#define HID     1024
#define M_DIM   (SLEN * BSZ)   // 32768

typedef __attribute__((ext_vector_type(16))) __bf16 v16bf;
typedef __attribute__((ext_vector_type(8)))  __bf16 v8bf;
typedef __attribute__((ext_vector_type(4)))  __bf16 v4bf;
typedef __attribute__((ext_vector_type(8)))  float  v8f;

union AFrag { v16bf v; v8bf h[2]; };

// ---------------------------------------------------------------------------
// f32 -> bf16 conversion (vectorized x4)
// ---------------------------------------------------------------------------
__global__ __launch_bounds__(256) void cvt_bf16_kernel(const float* __restrict__ src,
                                                       __bf16* __restrict__ dst,
                                                       int n4) {
    int i      = blockIdx.x * blockDim.x + threadIdx.x;
    int stride = gridDim.x * blockDim.x;
    const float4* s4 = (const float4*)src;
    v4bf* d4 = (v4bf*)dst;
    for (; i < n4; i += stride) {
        float4 v = s4[i];
        v4bf o;
        o[0] = (__bf16)v.x;
        o[1] = (__bf16)v.y;
        o[2] = (__bf16)v.z;
        o[3] = (__bf16)v.w;
        d4[i] = o;
    }
}

// ---------------------------------------------------------------------------
// Dual GEMM: out_z = X @ Wz^T, out_f = X @ Wf^T.
// X:  [M=32768, K=1024] bf16 row-major
// W*: [N=1024,  K=1024] bf16 row-major (K-major == what WMMA's B wants per-col)
// One wave -> 32(M) x 32(N) tile for BOTH z and f:
//   2 A fragments (rows m0..m0+15, m0+16..m0+31) x 4 B fragments (z/f x 2 N)
//   -> 8 WMMAs per K=32 chunk vs 12 b128 loads.
// ---------------------------------------------------------------------------
__global__ __launch_bounds__(256) void gemm_zf_kernel(const __bf16* __restrict__ xb,
                                                      const __bf16* __restrict__ wz,
                                                      const __bf16* __restrict__ wf,
                                                      float* __restrict__ oz,
                                                      float* __restrict__ of_) {
    const int lane   = threadIdx.x & 31;
    const int waveId = blockIdx.x * (blockDim.x >> 5) + (threadIdx.x >> 5);

    const int nStrips = HID / 32;            // 32 strips of width 32
    const int mTile   = waveId / nStrips;    // 0..1023  (32 rows each)
    const int nStrip  = waveId % nStrips;    // 0..31
    const int m0 = mTile * 32;
    const int n0 = nStrip * 32;

    const int r   = lane & 15;
    // A (16x32 bf16) per-lane K sub-offset: lanes 0-15 -> K 0..15,
    // lanes 16-31 -> K 16..31 of row M = lane&15.
    const int hiA = (lane >> 4) ? 16 : 0;
    // B (32x16 bf16): lane = column N, lanes 0-15 hold K 0..15, lanes 16-31 K 16..31.
    const int hiB = (lane >> 4) ? 16 : 0;

    v8f accz00 = {}, accz01 = {}, accz10 = {}, accz11 = {};
    v8f accf00 = {}, accf01 = {}, accf10 = {}, accf11 = {};

    const __bf16* pa0 = xb + (size_t)(m0 + r) * IN_DIM + hiA;         // rows m0..m0+15
    const __bf16* pa1 = xb + (size_t)(m0 + 16 + r) * IN_DIM + hiA;    // rows m0+16..m0+31
    const __bf16* pz0 = wz + (size_t)(n0 + r) * IN_DIM + hiB;
    const __bf16* pz1 = wz + (size_t)(n0 + 16 + r) * IN_DIM + hiB;
    const __bf16* pf0 = wf + (size_t)(n0 + r) * IN_DIM + hiB;
    const __bf16* pf1 = wf + (size_t)(n0 + 16 + r) * IN_DIM + hiB;

    for (int kk = 0; kk < IN_DIM; kk += 32) {
        AFrag a0, a1;
        a0.h[0] = *(const v8bf*)(pa0 + kk);
        a0.h[1] = *(const v8bf*)(pa0 + kk + 8);
        a1.h[0] = *(const v8bf*)(pa1 + kk);
        a1.h[1] = *(const v8bf*)(pa1 + kk + 8);

        AFrag bz0, bz1, bf0, bf1;
        bz0.h[0] = *(const v8bf*)(pz0 + kk);
        bz0.h[1] = *(const v8bf*)(pz0 + kk + 8);
        bz1.h[0] = *(const v8bf*)(pz1 + kk);
        bz1.h[1] = *(const v8bf*)(pz1 + kk + 8);
        bf0.h[0] = *(const v8bf*)(pf0 + kk);
        bf0.h[1] = *(const v8bf*)(pf0 + kk + 8);
        bf1.h[0] = *(const v8bf*)(pf1 + kk);
        bf1.h[1] = *(const v8bf*)(pf1 + kk + 8);

        accz00 = __builtin_amdgcn_wmma_f32_16x16x32_bf16(false, a0.v, false, bz0.v,
                                                         (short)0, accz00, false, false);
        accz01 = __builtin_amdgcn_wmma_f32_16x16x32_bf16(false, a1.v, false, bz0.v,
                                                         (short)0, accz01, false, false);
        accz10 = __builtin_amdgcn_wmma_f32_16x16x32_bf16(false, a0.v, false, bz1.v,
                                                         (short)0, accz10, false, false);
        accz11 = __builtin_amdgcn_wmma_f32_16x16x32_bf16(false, a1.v, false, bz1.v,
                                                         (short)0, accz11, false, false);
        accf00 = __builtin_amdgcn_wmma_f32_16x16x32_bf16(false, a0.v, false, bf0.v,
                                                         (short)0, accf00, false, false);
        accf01 = __builtin_amdgcn_wmma_f32_16x16x32_bf16(false, a1.v, false, bf0.v,
                                                         (short)0, accf01, false, false);
        accf10 = __builtin_amdgcn_wmma_f32_16x16x32_bf16(false, a0.v, false, bf1.v,
                                                         (short)0, accf10, false, false);
        accf11 = __builtin_amdgcn_wmma_f32_16x16x32_bf16(false, a1.v, false, bf1.v,
                                                         (short)0, accf11, false, false);
    }

    // C/D f32 16x16 layout: lanes 0-15 VGPR j -> (M=j, N=lane);
    // lanes 16-31 VGPR j -> (M=8+j, N=lane-16).
    const int n    = lane & 15;
    const int mrow = (lane >> 4) * 8;
    float* poz0 = oz  + (size_t)(m0 + mrow) * HID + (n0 + n);        // rows m0..
    float* pof0 = of_ + (size_t)(m0 + mrow) * HID + (n0 + n);
    float* poz1 = oz  + (size_t)(m0 + 16 + mrow) * HID + (n0 + n);   // rows m0+16..
    float* pof1 = of_ + (size_t)(m0 + 16 + mrow) * HID + (n0 + n);
#pragma unroll
    for (int j = 0; j < 8; ++j) {
        poz0[(size_t)j * HID]      = accz00[j];
        poz0[(size_t)j * HID + 16] = accz10[j];
        poz1[(size_t)j * HID]      = accz01[j];
        poz1[(size_t)j * HID + 16] = accz11[j];
        pof0[(size_t)j * HID]      = accf00[j];
        pof0[(size_t)j * HID + 16] = accf10[j];
        pof1[(size_t)j * HID]      = accf01[j];
        pof1[(size_t)j * HID + 16] = accf11[j];
    }
}

// ---------------------------------------------------------------------------
// Diagonal recurrence: one thread per (b,h) channel; 32768 threads.
// Coalesced in h; software-pipelined next-step loads.
// ---------------------------------------------------------------------------
__global__ __launch_bounds__(256) void scan_kernel(const float* __restrict__ oz,
                                                   const float* __restrict__ of_,
                                                   const float* __restrict__ state,
                                                   const float* __restrict__ wvz,
                                                   const float* __restrict__ wvf,
                                                   const float* __restrict__ bz,
                                                   const float* __restrict__ bf_,
                                                   float* __restrict__ out) {
    const int tid = blockIdx.x * blockDim.x + threadIdx.x;   // b*HID + h
    const int h   = tid & (HID - 1);

    const float wz  = wvz[h];
    const float wf  = wvf[h];
    const float cbz = bz[h];
    const float cbf = bf_[h];

    float cell = state[tid];

    size_t idx = (size_t)tid;
    float z = oz[idx];
    float f = of_[idx];

    for (int s = 0; s < SLEN; ++s) {
        const size_t nidx = idx + (size_t)(BSZ * HID);
        float zn = 0.0f, fn = 0.0f;
        if (s + 1 < SLEN) { zn = oz[nidx]; fn = of_[nidx]; }  // prefetch next step

        const float zpre = z + wz * cell + cbz;
        const float fpre = f + wf * cell + cbf;

        const float e2 = __expf(2.0f * zpre);
        const float zp = (e2 - 1.0f) / (e2 + 1.0f);
        const float fp = 1.0f / (1.0f + __expf(-fpre));

        cell = cell * fp + (1.0f - fp) * zp;
        out[idx] = cell;

        idx = nidx; z = zn; f = fn;
    }
}

// ---------------------------------------------------------------------------
// Launch
// ---------------------------------------------------------------------------
extern "C" void kernel_launch(void* const* d_in, const int* in_sizes, int n_in,
                              void* d_out, int out_size, void* d_ws, size_t ws_size,
                              hipStream_t stream) {
    const float* x      = (const float*)d_in[0];
    const float* state  = (const float*)d_in[1];
    const float* wm_z   = (const float*)d_in[2];
    const float* wm_f   = (const float*)d_in[3];
    const float* wv_z   = (const float*)d_in[4];
    const float* wv_f   = (const float*)d_in[5];
    const float* bias_z = (const float*)d_in[6];
    const float* bias_f = (const float*)d_in[7];
    float* out = (float*)d_out;

    char* ws = (char*)d_ws;
    // workspace layout (bytes):
    //   [0,      64MB)  x_bf16      (32768*1024*2)
    //   [64MB,   66MB)  wm_z bf16   (1024*1024*2)
    //   [66MB,   68MB)  wm_f bf16
    //   [68MB,  196MB)  out_z f32   (32768*1024*4)
    //   [196MB, 324MB)  out_f f32
    __bf16* xb  = (__bf16*)(ws);
    __bf16* wzb = (__bf16*)(ws + (size_t)67108864);
    __bf16* wfb = (__bf16*)(ws + (size_t)69206016);
    float*  oz  = (float*) (ws + (size_t)71303168);
    float*  of_ = (float*) (ws + (size_t)71303168 + (size_t)134217728);

    // 1) f32 -> bf16 conversions
    cvt_bf16_kernel<<<8192, 256, 0, stream>>>(x,    xb,  (M_DIM * IN_DIM) / 4);
    cvt_bf16_kernel<<<1024, 256, 0, stream>>>(wm_z, wzb, (HID * IN_DIM) / 4);
    cvt_bf16_kernel<<<1024, 256, 0, stream>>>(wm_f, wfb, (HID * IN_DIM) / 4);

    // 2) dual GEMM: 1024 mTiles(32 rows) * 32 nStrips = 32768 waves, 8 waves/block
    gemm_zf_kernel<<<4096, 256, 0, stream>>>(xb, wzb, wfb, oz, of_);

    // 3) recurrence scan: 32768 channels
    scan_kernel<<<(BSZ * HID) / 256, 256, 0, stream>>>(oz, of_, state,
                                                       wv_z, wv_f, bias_z, bias_f,
                                                       out);
}
